// Attention_30202210025654
// MI455X (gfx1250) — matrix operand order repaired and checked by reading, not compiled, not run on hardware
//
#include <hip/hip_runtime.h>
#include <hip/hip_bf16.h>
#include <stdint.h>

typedef __bf16 bf16;
typedef __bf16 bf16x16 __attribute__((ext_vector_type(16)));
typedef float  f32x8   __attribute__((ext_vector_type(8)));
typedef unsigned int u32x4 __attribute__((ext_vector_type(4)));
typedef int          i32x4 __attribute__((ext_vector_type(4)));

#define B_    64
#define N_    197
#define NPAD  224          // keys/queries padded to 7 * 32
#define DIM_  768
#define H_    12
#define KD_   64
#define D_    128
#define DH_   1536
#define HQKV_ 3072
#define ROWS_ (B_ * N_)    // 12608 = 64 * 197
#define EPS_  1e-5f
#define SCALE_ 0.125f      // 64^-0.5

union FragBF { bf16x16 v; u32x4 u[2]; };

// ---- CDNA5 async global->LDS staging (ASYNCcnt path), with safe fallback ----
#if defined(__AMDGCN__) && __has_builtin(__builtin_amdgcn_global_load_async_to_lds_b128)
#define HAVE_ASYNC_LDS 1
#else
#define HAVE_ASYNC_LDS 0
#endif

#define AS1_ __attribute__((address_space(1)))
#define AS3_ __attribute__((address_space(3)))

__device__ __forceinline__ void copy16_g2l(bf16* lds, const bf16* g) {
#if HAVE_ASYNC_LDS
  __builtin_amdgcn_global_load_async_to_lds_b128(
      (AS1_ i32x4*)g, (AS3_ i32x4*)lds, 0, 0);
#else
  *(u32x4*)lds = *(const u32x4*)g;
#endif
}

__device__ __forceinline__ void wait_async_lds() {
#if HAVE_ASYNC_LDS
#if __has_builtin(__builtin_amdgcn_s_wait_asynccnt)
  __builtin_amdgcn_s_wait_asynccnt(0);
#else
  asm volatile("s_wait_asynccnt 0x0" ::: "memory");
#endif
#endif
}

// ---------------------------------------------------------------- utilities
__global__ void k_zero_f32(float* p, int n) {
  int i = blockIdx.x * 256 + threadIdx.x;
  if (i < n) p[i] = 0.f;
}

__global__ void k_cvt_bf16(const float* __restrict__ in, bf16* __restrict__ out, int n) {
  int i = blockIdx.x * 256 + threadIdx.x;
  if (i < n) out[i] = (bf16)in[i];
}

// --------------------------------------------------- tiled NT GEMM (bf16 WMMA)
// C[M,Nt] = A[M,K] * Bm[Nt,K]^T, f32 accumulate. 128x128 tile, K step 64
// (two WMMA K-steps per LDS stage -> 16 v_wmma per barrier pair).
__global__ __launch_bounds__(256)
void k_gemm_nt_bf16(const bf16* __restrict__ A, const bf16* __restrict__ Bm,
                    float* __restrict__ C, int M, int Nt, int K)
{
  __shared__ bf16 As[128][72];   // 64 K elems + 8 pad (row stride 144B, 16B aligned)
  __shared__ bf16 Bs[128][72];

  const int tid  = threadIdx.x;
  const int wid  = tid >> 5;           // wave32
  const int lane = tid & 31;
  const int l15  = lane & 15;
  const int kh   = lane >> 4;
  const int mbase = blockIdx.x * 128;
  const int nbase = blockIdx.y * 128;
  const int wm = (wid & 1) * 64;       // 2x4 wave grid -> 64x32 per wave
  const int wn = (wid >> 1) * 32;
  const bool fullM = (mbase + 128 <= M);

  f32x8 acc[4][2];
  for (int i = 0; i < 4; ++i)
    for (int j = 0; j < 2; ++j)
      for (int e = 0; e < 8; ++e) acc[i][j][e] = 0.f;

  for (int k0 = 0; k0 < K; k0 += 64) {
    __syncthreads();
    // stage A tile (128x64) and B tile (128x64); 4 b128 chunks per thread each
    if (fullM) {
#pragma unroll
      for (int c = tid; c < 1024; c += 256) {
        int row = c >> 3, col8 = (c & 7) * 8;
        copy16_g2l(&As[row][col8], A + (size_t)(mbase + row) * K + k0 + col8);
      }
    } else {
#pragma unroll
      for (int c = tid; c < 1024; c += 256) {
        int row = c >> 3, col8 = (c & 7) * 8;
        int gr = mbase + row;
        u32x4 d = {0, 0, 0, 0};
        if (gr < M) d = *(const u32x4*)(A + (size_t)gr * K + k0 + col8);
        *(u32x4*)(&As[row][col8]) = d;
      }
    }
#pragma unroll
    for (int c = tid; c < 1024; c += 256) {
      int row = c >> 3, col8 = (c & 7) * 8;       // Nt is a multiple of 128
      copy16_g2l(&Bs[row][col8], Bm + (size_t)(nbase + row) * K + k0 + col8);
    }
    wait_async_lds();
    __syncthreads();

#pragma unroll
    for (int ks = 0; ks < 64; ks += 32) {
      FragBF bfr[2];
#pragma unroll
      for (int ni = 0; ni < 2; ++ni) {           // B: lane = column, 16 contig K
        int col  = wn + ni * 16 + l15;
        int koff = ks + kh * 16;
        bfr[ni].u[0] = *(const u32x4*)(&Bs[col][koff]);
        bfr[ni].u[1] = *(const u32x4*)(&Bs[col][koff + 8]);
      }
#pragma unroll
      for (int mi = 0; mi < 4; ++mi) {           // A: lane = row, 16 contig K
        FragBF a;
        int row  = wm + mi * 16 + l15;
        int koff = ks + kh * 16;
        a.u[0] = *(const u32x4*)(&As[row][koff]);
        a.u[1] = *(const u32x4*)(&As[row][koff + 8]);
#pragma unroll
        for (int ni = 0; ni < 2; ++ni) {
          acc[mi][ni] = __builtin_amdgcn_wmma_f32_16x16x32_bf16(
              false, a.v, false, bfr[ni].v, (short)0, acc[mi][ni], false, false);
        }
      }
    }
  }

  const int rowadd = (lane >= 16) ? 8 : 0;
  if (fullM) {
#pragma unroll
    for (int mi = 0; mi < 4; ++mi)
      for (int ni = 0; ni < 2; ++ni)
        for (int r = 0; r < 8; ++r) {
          int gr = mbase + wm + mi * 16 + r + rowadd;
          C[(size_t)gr * Nt + nbase + wn + ni * 16 + l15] = acc[mi][ni][r];
        }
  } else {
    for (int mi = 0; mi < 4; ++mi)
      for (int ni = 0; ni < 2; ++ni)
        for (int r = 0; r < 8; ++r) {
          int gr = mbase + wm + mi * 16 + r + rowadd;
          if (gr < M)
            C[(size_t)gr * Nt + nbase + wn + ni * 16 + l15] = acc[mi][ni][r];
        }
  }
}

// ----------------------------------------------------- BN statistics (train)
__global__ void k_bn_partial(const float* __restrict__ X, int F,
                             float* __restrict__ sum, float* __restrict__ sumsq)
{
  int f  = blockIdx.x * 256 + threadIdx.x;   // F is a multiple of 256
  int r0 = blockIdx.y * 197;                 // 12608 = 64 * 197
  float s = 0.f, s2 = 0.f;
  for (int r = r0; r < r0 + 197; ++r) {
    float v = X[(size_t)r * F + f];
    s += v; s2 += v * v;
  }
  atomicAdd(&sum[f], s);
  atomicAdd(&sumsq[f], s2);
}

__global__ void k_bn_finalize(const float* __restrict__ sum, const float* __restrict__ sumsq,
                              const float* __restrict__ g, const float* __restrict__ b,
                              float* __restrict__ scale, float* __restrict__ shift, int F)
{
  int f = blockIdx.x * 256 + threadIdx.x;
  if (f >= F) return;
  const float inv = 1.f / (float)ROWS_;
  float m   = sum[f] * inv;
  float var = sumsq[f] * inv - m * m;      // biased variance, matches reference
  float sc  = g[f] * rsqrtf(var + EPS_);
  scale[f] = sc;
  shift[f] = b[f] - m * sc;
}

// ------------------------- BN apply + head split; q scaled; k row-major; v^T
__global__ void k_qkv_scatter(const float* __restrict__ raw,
                              const float* __restrict__ scale, const float* __restrict__ shift,
                              bf16* __restrict__ q, bf16* __restrict__ k, bf16* __restrict__ vT)
{
  int b = blockIdx.x / NPAD;
  int n = blockIdx.x % NPAD;     // includes zero padding rows 197..223
  for (int it = 0; it < HQKV_ / 256; ++it) {
    int f = it * 256 + threadIdx.x;
    int h = f >> 8, j = f & 255;               // per-head block = 2*KD + D = 256
    float val = 0.f;
    if (n < N_) val = raw[(size_t)(b * N_ + n) * HQKV_ + f] * scale[f] + shift[f];
    int bh = b * H_ + h;
    if (j < KD_)
      q[((size_t)bh * NPAD + n) * KD_ + j] = (bf16)(val * SCALE_);
    else if (j < 2 * KD_)
      k[((size_t)bh * NPAD + n) * KD_ + (j - KD_)] = (bf16)val;
    else
      vT[((size_t)bh * D_ + (j - 2 * KD_)) * NPAD + n] = (bf16)val;
  }
}

// --------------- fused attention: S=QK^T (WMMA), softmax, cls, O=PV (WMMA),
//                 Hardswish epilogue -> bf16 activations for proj GEMM
__global__ __launch_bounds__(256)
void k_attention(const bf16* __restrict__ q, const bf16* __restrict__ kmat,
                 const bf16* __restrict__ vT, bf16* __restrict__ hs,
                 float* __restrict__ cls)
{
  __shared__ bf16  Qs[32][72];     // 32-row q tile, KD=64, padded stride
  __shared__ float Ss[32][224];    // scores
  __shared__ bf16  Ps[32][224];    // probabilities

  const int tid  = threadIdx.x;
  const int wid  = tid >> 5;
  const int lane = tid & 31;
  const int l15  = lane & 15;
  const int kh   = lane >> 4;
  const int qt = blockIdx.x;       // 7 q-tiles of 32 rows
  const int h  = blockIdx.y;
  const int b  = blockIdx.z;
  const int bh = b * H_ + h;

  { // load q tile (pre-scaled, padded buffer: no guards needed)
    int row = tid >> 3, col8 = (tid & 7) * 8;
    copy16_g2l(&Qs[row][col8],
               q + ((size_t)bh * NPAD + qt * 32 + row) * KD_ + col8);
    wait_async_lds();
  }
  __syncthreads();

  // S = Q K^T : 2 x 14 tiles of 16x16 over 8 waves
  for (int t = wid; t < 28; t += 8) {
    int mt = t & 1, nt = t >> 1;
    f32x8 acc; for (int e = 0; e < 8; ++e) acc[e] = 0.f;
#pragma unroll
    for (int ks = 0; ks < KD_; ks += 32) {
      FragBF a, bb;
      int arow = mt * 16 + l15;
      a.u[0] = *(const u32x4*)&Qs[arow][ks + kh * 16];
      a.u[1] = *(const u32x4*)&Qs[arow][ks + kh * 16 + 8];
      int key = nt * 16 + l15;     // B fragment: lane = key column, contig KD
      const bf16* kp = kmat + ((size_t)bh * NPAD + key) * KD_ + ks + kh * 16;
      bb.u[0] = *(const u32x4*)kp;
      bb.u[1] = *(const u32x4*)(kp + 8);
      acc = __builtin_amdgcn_wmma_f32_16x16x32_bf16(
          false, a.v, false, bb.v, (short)0, acc, false, false);
    }
    int rowadd = (lane >= 16) ? 8 : 0;
    int key = nt * 16 + l15;
#pragma unroll
    for (int r = 0; r < 8; ++r)
      Ss[mt * 16 + r + rowadd][key] = (key < N_) ? acc[r] : -1e30f;
  }
  __syncthreads();

  // row softmax: 8 lanes per row, wave-local shuffles for the reduction
  {
    int row = tid >> 3, l8 = tid & 7;
    float mx = -1e30f;
    for (int j = l8; j < NPAD; j += 8) mx = fmaxf(mx, Ss[row][j]);
    mx = fmaxf(mx, __shfl_xor(mx, 1));
    mx = fmaxf(mx, __shfl_xor(mx, 2));
    mx = fmaxf(mx, __shfl_xor(mx, 4));
    float sm = 0.f;
    for (int j = l8; j < NPAD; j += 8) {
      float e = __expf(Ss[row][j] - mx);
      Ss[row][j] = e; sm += e;
    }
    sm += __shfl_xor(sm, 1);
    sm += __shfl_xor(sm, 2);
    sm += __shfl_xor(sm, 4);
    float inv = 1.f / sm;
    bool isCls = (qt == 0) && (row == 0);   // global q row 0
    for (int j = l8; j < NPAD; j += 8) {
      float p = Ss[row][j] * inv;
      Ps[row][j] = (bf16)p;
      if (isCls && j >= 1 && j < N_)
        atomicAdd(&cls[b * (N_ - 1) + (j - 1)], p * (1.f / H_));
    }
  }
  __syncthreads();

  // O = P V : 2 x 8 tiles of 16x16; K = 224 padded keys (zero probs/v beyond 197)
  for (int t = wid; t < 16; t += 8) {
    int mt = t & 1, nt = t >> 1;
    f32x8 acc; for (int e = 0; e < 8; ++e) acc[e] = 0.f;
#pragma unroll
    for (int ks = 0; ks < NPAD; ks += 32) {
      FragBF a, bb;
      int arow = mt * 16 + l15;
      a.u[0] = *(const u32x4*)&Ps[arow][ks + kh * 16];
      a.u[1] = *(const u32x4*)&Ps[arow][ks + kh * 16 + 8];
      int d = nt * 16 + l15;       // B fragment: lane = d column, contig keys (v^T)
      const bf16* vp = vT + ((size_t)bh * D_ + d) * NPAD + ks + kh * 16;
      bb.u[0] = *(const u32x4*)vp;
      bb.u[1] = *(const u32x4*)(vp + 8);
      acc = __builtin_amdgcn_wmma_f32_16x16x32_bf16(
          false, a.v, false, bb.v, (short)0, acc, false, false);
    }
    int rowadd = (lane >= 16) ? 8 : 0;
    int d = nt * 16 + l15;
#pragma unroll
    for (int r = 0; r < 8; ++r) {
      int qrow = qt * 32 + mt * 16 + r + rowadd;
      if (qrow < N_) {
        float x = acc[r];
        float hsw = x * fminf(fmaxf(x + 3.f, 0.f), 6.f) * (1.f / 6.f);
        hs[(size_t)(b * N_ + qrow) * DH_ + h * D_ + d] = (bf16)hsw;
      }
    }
  }
}

__global__ void k_bn_apply(const float* __restrict__ yr,
                           const float* __restrict__ scale, const float* __restrict__ shift,
                           float* __restrict__ out)
{
  size_t i = (size_t)blockIdx.x * 256 + threadIdx.x;
  if (i >= (size_t)ROWS_ * DIM_) return;
  int c = (int)(i % DIM_);
  out[i] = yr[i] * scale[c] + shift[c];
}

// ---------------------------------------------------------------- launcher
extern "C" void kernel_launch(void* const* d_in, const int* in_sizes, int n_in,
                              void* d_out, int out_size, void* d_ws, size_t ws_size,
                              hipStream_t stream)
{
  (void)in_sizes; (void)n_in; (void)out_size; (void)ws_size;
  const float* x      = (const float*)d_in[0];
  const float* qkv_w  = (const float*)d_in[1];
  const float* qkv_g  = (const float*)d_in[2];
  const float* qkv_b  = (const float*)d_in[3];
  const float* proj_w = (const float*)d_in[4];
  const float* proj_g = (const float*)d_in[5];
  const float* proj_b = (const float*)d_in[6];

  float* y_out   = (float*)d_out;
  float* cls_out = y_out + (size_t)ROWS_ * DIM_;

  char* ws = (char*)d_ws;
  size_t off = 0;
  auto take = [&](size_t bytes) -> char* {
    char* p = ws + off;
    off = (off + bytes + 255) & ~(size_t)255;
    return p;
  };

  bf16*  x_bf    = (bf16*)take((size_t)ROWS_ * DIM_ * 2);
  bf16*  wq_bf   = (bf16*)take((size_t)HQKV_ * DIM_ * 2);
  bf16*  wp_bf   = (bf16*)take((size_t)DIM_ * DH_ * 2);
  float* stats_q = (float*)take((size_t)4 * HQKV_ * 4);   // sum, sumsq, scale, shift
  float* stats_p = (float*)take((size_t)4 * DIM_ * 4);
  bf16*  qbuf    = (bf16*)take((size_t)B_ * H_ * NPAD * KD_ * 2);
  bf16*  kbuf    = (bf16*)take((size_t)B_ * H_ * NPAD * KD_ * 2);
  bf16*  vTbuf   = (bf16*)take((size_t)B_ * H_ * D_ * NPAD * 2);
  char*  rawblk  = take((size_t)ROWS_ * HQKV_ * 4);       // 155 MB, reused below
  float* raw     = (float*)rawblk;
  bf16*  hsbuf   = (bf16*)rawblk;                          // alias (raw dead after scatter)
  float* yraw    = (float*)(rawblk + (((size_t)ROWS_ * DH_ * 2 + 255) & ~(size_t)255));

  float* qs_sum = stats_q,           *qs_sq = stats_q + HQKV_;
  float* qs_sc  = stats_q + 2*HQKV_, *qs_sh = stats_q + 3*HQKV_;
  float* ps_sum = stats_p,           *ps_sq = stats_p + DIM_;
  float* ps_sc  = stats_p + 2*DIM_,  *ps_sh = stats_p + 3*DIM_;

  // zero accumulators + cls output region (d_out is poisoned by harness)
  k_zero_f32<<<(4 * HQKV_ + 255) / 256, 256, 0, stream>>>(stats_q, 4 * HQKV_);
  k_zero_f32<<<(4 * DIM_ + 255) / 256, 256, 0, stream>>>(stats_p, 4 * DIM_);
  k_zero_f32<<<(B_ * (N_ - 1) + 255) / 256, 256, 0, stream>>>(cls_out, B_ * (N_ - 1));

  // f32 -> bf16 conversions (one bandwidth pass each)
  k_cvt_bf16<<<(ROWS_ * DIM_ + 255) / 256, 256, 0, stream>>>(x, x_bf, ROWS_ * DIM_);
  k_cvt_bf16<<<(HQKV_ * DIM_ + 255) / 256, 256, 0, stream>>>(qkv_w, wq_bf, HQKV_ * DIM_);
  k_cvt_bf16<<<(DIM_ * DH_ + 255) / 256, 256, 0, stream>>>(proj_w, wp_bf, DIM_ * DH_);

  // QKV GEMM: [12608,768] x [3072,768]^T -> [12608,3072]
  dim3 g1((ROWS_ + 127) / 128, HQKV_ / 128);
  k_gemm_nt_bf16<<<g1, 256, 0, stream>>>(x_bf, wq_bf, raw, ROWS_, HQKV_, DIM_);

  // BN stats + folded scale/shift
  dim3 g2(HQKV_ / 256, 64);
  k_bn_partial<<<g2, 256, 0, stream>>>(raw, HQKV_, qs_sum, qs_sq);
  k_bn_finalize<<<HQKV_ / 256, 256, 0, stream>>>(qs_sum, qs_sq, qkv_g, qkv_b, qs_sc, qs_sh, HQKV_);

  // BN apply + split into q (scaled), k, v^T bf16 with key padding to 224
  k_qkv_scatter<<<B_ * NPAD, 256, 0, stream>>>(raw, qs_sc, qs_sh, qbuf, kbuf, vTbuf);

  // fused attention (+cls mean, +Hardswish) -> bf16 activations
  dim3 ga(NPAD / 32, H_, B_);
  k_attention<<<ga, 256, 0, stream>>>(qbuf, kbuf, vTbuf, hsbuf, cls_out);

  // proj GEMM: [12608,1536] x [768,1536]^T -> [12608,768]
  dim3 g3((ROWS_ + 127) / 128, DIM_ / 128);
  k_gemm_nt_bf16<<<g3, 256, 0, stream>>>(hsbuf, wp_bf, yraw, ROWS_, DIM_, DH_);

  // final BN -> d_out
  dim3 g4(DIM_ / 256, 64);
  k_bn_partial<<<g4, 256, 0, stream>>>(yraw, DIM_, ps_sum, ps_sq);
  k_bn_finalize<<<DIM_ / 256, 256, 0, stream>>>(ps_sum, ps_sq, proj_g, proj_b, ps_sc, ps_sh, DIM_);
  k_bn_apply<<<(int)(((size_t)ROWS_ * DIM_ + 255) / 256), 256, 0, stream>>>(yraw, ps_sc, ps_sh, y_out);
}
